// GNN_82729660055705
// MI455X (gfx1250) — matrix operand order repaired and checked by reading, not compiled, not run on hardware
//
#include <hip/hip_runtime.h>
#include <hip/hip_bf16.h>
#include <math.h>

#define NNODE 50000
#define NEDGE 500000
#define FW 128
#define TROWS 35                 // 34 projected table rows + 1 scaled-bias row
#define TSZ (3 * TROWS * FW)     // 13440 floats
#define EPB 160                  // edges per block (8 waves x 20)

typedef __attribute__((ext_vector_type(16))) _Float16 v16h;
typedef __attribute__((ext_vector_type(8)))  float    v8f;

__device__ __forceinline__ float gelu_f(float v) {
    return 0.5f * v * (1.0f + erff(v * 0.70710678118654752f));
}

// ---- WMMA fragment loaders (16x16x32 f16, wave32 layouts per CDNA5 ISA) ----
// A: row-major [*, 128]; lane<16: row=lane, K = k0+{0..7,16..23}; lane>=16: K = k0+{8..15,24..31}
__device__ __forceinline__ v16h load_a_frag(const _Float16* __restrict__ A, int row, int k0, int hs) {
    v16h a;
    const _Float16* p = A + row * FW + k0 + hs * 8;
#pragma unroll
    for (int i = 0; i < 8; ++i) a[i] = p[i];
#pragma unroll
    for (int i = 0; i < 8; ++i) a[8 + i] = p[16 + i];
    return a;
}
// B: weights stored col-major (Wt[col*128 + k]); lane<16: col=lane, K=k0+{0..7,16..23}; lane>=16: K=k0+{8..15,24..31}
__device__ __forceinline__ v16h load_b_frag(const _Float16* __restrict__ Wt, int col, int k0, int hs) {
    v16h b;
    const _Float16* p = Wt + col * FW + k0 + hs * 8;
#pragma unroll
    for (int i = 0; i < 8; ++i) b[i] = p[i];
#pragma unroll
    for (int i = 0; i < 8; ++i) b[8 + i] = p[16 + i];
    return b;
}

// ---- small converters ----
__global__ void k_cvt_w(const float* __restrict__ w, _Float16* __restrict__ wt) {
    int i = blockIdx.x * 256 + threadIdx.x;   // 16384 threads, i = k*128 + c
    int c = i & 127, k = i >> 7;
    wt[c * FW + k] = (_Float16)w[i];          // transpose to col-major
}

__global__ void k_cvt_x(const float* __restrict__ x, _Float16* __restrict__ xh, int n) {
    int i = blockIdx.x * 256 + threadIdx.x;
    if (i < n) xh[i] = (_Float16)x[i];
}

// ---- project embedding tables through remix weights, fold xw + scale + bias ----
// T layout: [3][35][128]; rows 0..33 = projected table rows (offsets 0,6,13,16), row 34 = scaled bias
__global__ void k_tables(const float* r0w, const float* r0b, const float* r1w, const float* r1b,
                         const float* r2w, const float* r2b,
                         const float* e0, const float* e1, const float* e2, const float* e3,
                         const float* einit, const float* init0, float* __restrict__ T) {
    int col = threadIdx.x;  // 128
    float ew0 = expf(einit[0]), ew1 = expf(einit[1]), ew2 = expf(einit[2]), ew3 = expf(einit[3]);
    float inv = rsqrtf(ew0 + ew1 + ew2 + ew3);
    float xw[4] = { ew0 * inv, ew1 * inv, ew2 * inv, ew3 * inv };
    float sc_e = expf(init0[2]);
    float sc_v = expf(init0[3]);
    float scale[3] = { sc_e, sc_e, sc_v };
    const float* rw[3] = { r0w, r1w, r2w };
    const float* rb[3] = { r0b, r1b, r2b };
    const float* emb[4] = { e0, e1, e2, e3 };
    const int rows[4] = { 6, 7, 3, 18 };
    const int off[4]  = { 0, 6, 13, 16 };
    for (int j = 0; j < 3; ++j) {
        float* Tj = T + j * TROWS * FW;
        for (int c = 0; c < 4; ++c) {
            for (int r = 0; r < rows[c]; ++r) {
                float s = 0.f;
                for (int k = 0; k < 64; ++k) s += emb[c][r * 64 + k] * rw[j][k * FW + col];
                Tj[(off[c] + r) * FW + col] = s * xw[c] * scale[j];
            }
        }
        Tj[34 * FW + col] = rb[j][col] * scale[j];
    }
}

// ---- pre: xx = LN(x @ pre_w + pre_b), output f16 ----
__global__ void k_pre(const _Float16* __restrict__ xh, const _Float16* __restrict__ preWt,
                      const float* __restrict__ pre_b, _Float16* __restrict__ xx_h) {
    __shared__ float sm[16 * FW];
    int t = threadIdx.x, lane = t & 31, wave = t >> 5;
    int hs = lane >> 4, lr = lane & 15;
    int row0 = blockIdx.x * 16, col0 = wave * 16;
    v8f acc = {};
#pragma unroll
    for (int kk = 0; kk < 4; ++kk) {
        v16h a = load_a_frag(xh, row0 + lr, kk * 32, hs);
        v16h b = load_b_frag(preWt, col0 + lr, kk * 32, hs);
        acc = __builtin_amdgcn_wmma_f32_16x16x32_f16(false, a, false, b, (short)0, acc, false, false);
    }
    int col = col0 + lr;
    float bias = pre_b[col];
#pragma unroll
    for (int j = 0; j < 8; ++j) sm[(j + hs * 8) * FW + col] = acc[j] + bias;
    __syncthreads();
    // LayerNorm: lanes 0-15 of wave w handle row 2w, lanes 16-31 row 2w+1
    int lrow = wave * 2 + hs;
    const float* rp = sm + lrow * FW;
    float s = 0.f, s2 = 0.f;
#pragma unroll
    for (int i = 0; i < 8; ++i) { float v = rp[lr * 8 + i]; s += v; s2 += v * v; }
#pragma unroll
    for (int m = 1; m < 16; m <<= 1) { s += __shfl_xor(s, m, 16); s2 += __shfl_xor(s2, m, 16); }
    float mean = s * (1.f / 128.f);
    float var  = s2 * (1.f / 128.f) - mean * mean;
    float rstd = rsqrtf(var + 1e-5f);
    _Float16* op = xx_h + (row0 + lrow) * FW;
#pragma unroll
    for (int i = 0; i < 8; ++i) op[lr * 8 + i] = (_Float16)((rp[lr * 8 + i] - mean) * rstd);
}

// ---- fused node-level GEMMs: M0g = gelu(xx@msg0+b0) f32; Q/K/V' = xx@msg{1,2,3}+b f16 ----
__global__ void k_qkvm(const _Float16* __restrict__ xx_h, const _Float16* __restrict__ msgWt,
                       const float* __restrict__ b0, const float* __restrict__ b1,
                       const float* __restrict__ b2, const float* __restrict__ b3,
                       float* __restrict__ M0g, _Float16* __restrict__ Qh,
                       _Float16* __restrict__ Kh, _Float16* __restrict__ Vh) {
    int t = threadIdx.x, lane = t & 31, wave = t >> 5;
    int hs = lane >> 4, lr = lane & 15;
    int row0 = blockIdx.x * 16, col0 = wave * 16;
    v16h a[4];
#pragma unroll
    for (int kk = 0; kk < 4; ++kk) a[kk] = load_a_frag(xx_h, row0 + lr, kk * 32, hs);
    int col = col0 + lr;
#pragma unroll
    for (int m = 0; m < 4; ++m) {
        const _Float16* Wt = msgWt + m * FW * FW;
        v8f acc = {};
#pragma unroll
        for (int kk = 0; kk < 4; ++kk) {
            v16h b = load_b_frag(Wt, col, kk * 32, hs);
            acc = __builtin_amdgcn_wmma_f32_16x16x32_f16(false, a[kk], false, b, (short)0, acc, false, false);
        }
        float bb = (m == 0) ? b0[col] : (m == 1) ? b1[col] : (m == 2) ? b2[col] : b3[col];
#pragma unroll
        for (int j = 0; j < 8; ++j) {
            int row = row0 + j + hs * 8;
            float v = acc[j] + bb;
            if      (m == 0) M0g[row * FW + col] = gelu_f(v);
            else if (m == 1) Qh[row * FW + col] = (_Float16)v;
            else if (m == 2) Kh[row * FW + col] = (_Float16)v;
            else             Vh[row * FW + col] = (_Float16)v;
        }
    }
}

// ---- per-edge: LDS-cached tables, 160 edges/block (8 waves x 20), 4 cols/lane ----
__global__ void k_edge(const int* __restrict__ ei, const int* __restrict__ ea,
                       const float* __restrict__ T,
                       const _Float16* __restrict__ Qh, const _Float16* __restrict__ Kh,
                       const _Float16* __restrict__ Vh, const float* __restrict__ init0,
                       float* __restrict__ agg) {
    __shared__ float sT[TSZ];                 // 53.76 KB: all 3 projected tables + biases
    int t = threadIdx.x, lane = t & 31, wave = t >> 5;
    for (int i = t; i < TSZ; i += 256) sT[i] = T[i];
    __syncthreads();

    const float* S0 = sT;
    const float* S1 = sT + TROWS * FW;
    const float* S2 = sT + 2 * TROWS * FW;
    float i0 = init0[0], i1 = init0[1];
    int ebase = blockIdx.x * EPB;

    for (int it = 0; it < EPB / 8; ++it) {
        int e = ebase + it * 8 + wave;        // strided: 8 consecutive edges across waves
        int src = ei[e];
        int dst = ei[NEDGE + e];
        int ro0 = ea[e * 4 + 0] * FW;
        int ro1 = (6  + ea[e * 4 + 1]) * FW;
        int ro2 = (13 + ea[e * 4 + 2]) * FW;
        int ro3 = (16 + ea[e * 4 + 3]) * FW;

        float vpre[4];
        float s = 0.f;
#pragma unroll
        for (int i = 0; i < 4; ++i) {
            int col = lane * 4 + i;           // lanes 0-15 -> head 0, lanes 16-31 -> head 1
            float R0 = S0[ro0 + col] + S0[ro1 + col] + S0[ro2 + col] + S0[ro3 + col] + S0[34 * FW + col];
            float R1 = S1[ro0 + col] + S1[ro1 + col] + S1[ro2 + col] + S1[ro3 + col] + S1[34 * FW + col];
            float R2 = S2[ro0 + col] + S2[ro1 + col] + S2[ro2 + col] + S2[ro3 + col] + S2[34 * FW + col];
            float q = (float)Qh[dst * FW + col] + R0;
            float k = (float)Kh[src * FW + col] + R1;
            s += q * k;
            vpre[i] = (float)Vh[src * FW + col] + R2;
        }
#pragma unroll
        for (int m = 1; m < 16; m <<= 1) s += __shfl_xor(s, m, 16);  // per-head reduce
        float att = expf(s * 0.125f * i0 + i1);                      // /sqrt(64), affine, exp
#pragma unroll
        for (int i = 0; i < 4; ++i) {
            int col = lane * 4 + i;
            float msg = gelu_f(vpre[i]) * att;
            __hip_atomic_fetch_add(&agg[dst * FW + col], msg,
                                   __ATOMIC_RELAXED, __HIP_MEMORY_SCOPE_AGENT);
        }
    }
}

// ---- H = M0g + agg, to f16 for post GEMM ----
__global__ void k_hbuild(const float* __restrict__ M0g, const float* __restrict__ agg,
                         _Float16* __restrict__ Hh, int n) {
    int i = blockIdx.x * 256 + threadIdx.x;
    if (i < n) Hh[i] = (_Float16)(M0g[i] + agg[i]);
}

// ---- out = x + H @ post_w + post_b ----
__global__ void k_post(const _Float16* __restrict__ Hh, const _Float16* __restrict__ postWt,
                       const float* __restrict__ post_b, const float* __restrict__ x,
                       float* __restrict__ out) {
    int t = threadIdx.x, lane = t & 31, wave = t >> 5;
    int hs = lane >> 4, lr = lane & 15;
    int row0 = blockIdx.x * 16, col0 = wave * 16;
    v8f acc = {};
#pragma unroll
    for (int kk = 0; kk < 4; ++kk) {
        v16h a = load_a_frag(Hh, row0 + lr, kk * 32, hs);
        v16h b = load_b_frag(postWt, col0 + lr, kk * 32, hs);
        acc = __builtin_amdgcn_wmma_f32_16x16x32_f16(false, a, false, b, (short)0, acc, false, false);
    }
    int col = col0 + lr;
    float bias = post_b[col];
#pragma unroll
    for (int j = 0; j < 8; ++j) {
        int row = row0 + j + hs * 8;
        out[row * FW + col] = x[row * FW + col] + acc[j] + bias;
    }
}

extern "C" void kernel_launch(void* const* d_in, const int* in_sizes, int n_in,
                              void* d_out, int out_size, void* d_ws, size_t ws_size,
                              hipStream_t stream) {
    const float* x      = (const float*)d_in[0];
    const int*   ei     = (const int*)d_in[1];
    const int*   ea     = (const int*)d_in[2];
    const float* pre_w  = (const float*)d_in[3];
    const float* pre_b  = (const float*)d_in[4];
    const float* msg_w[4] = { (const float*)d_in[5], (const float*)d_in[7],
                              (const float*)d_in[9], (const float*)d_in[11] };
    const float* msg_b[4] = { (const float*)d_in[6], (const float*)d_in[8],
                              (const float*)d_in[10], (const float*)d_in[12] };
    const float* r_w[3] = { (const float*)d_in[13], (const float*)d_in[15], (const float*)d_in[17] };
    const float* r_b[3] = { (const float*)d_in[14], (const float*)d_in[16], (const float*)d_in[18] };
    const float* post_w = (const float*)d_in[19];
    const float* post_b = (const float*)d_in[20];
    const float* e0 = (const float*)d_in[21];
    const float* e1 = (const float*)d_in[22];
    const float* e2 = (const float*)d_in[23];
    const float* e3 = (const float*)d_in[24];
    const float* einit = (const float*)d_in[25];
    const float* init0 = (const float*)d_in[26];
    float* out = (float*)d_out;

    // workspace carve-up (~128 MB total)
    char* p = (char*)d_ws;
    auto take = [&](size_t bytes) { char* r = p; p += (bytes + 255) & ~(size_t)255; return (void*)r; };
    const size_t NF = (size_t)NNODE * FW;
    _Float16* x_h    = (_Float16*)take(NF * 2);
    _Float16* xx_h   = (_Float16*)take(NF * 2);
    _Float16* preWt  = (_Float16*)take((size_t)FW * FW * 2);
    _Float16* msgWt  = (_Float16*)take((size_t)4 * FW * FW * 2);
    _Float16* postWt = (_Float16*)take((size_t)FW * FW * 2);
    float*    T      = (float*)take((size_t)TSZ * 4);
    _Float16* Qh     = (_Float16*)take(NF * 2);
    _Float16* Kh     = (_Float16*)take(NF * 2);
    _Float16* Vh     = (_Float16*)take(NF * 2);
    float*    M0g    = (float*)take(NF * 4);
    float*    agg    = (float*)take(NF * 4);
    _Float16* Hh     = (_Float16*)take(NF * 2);

    // prep
    k_cvt_w<<<64, 256, 0, stream>>>(pre_w, preWt);
    for (int m = 0; m < 4; ++m)
        k_cvt_w<<<64, 256, 0, stream>>>(msg_w[m], msgWt + (size_t)m * FW * FW);
    k_cvt_w<<<64, 256, 0, stream>>>(post_w, postWt);
    k_cvt_x<<<(int)((NF + 255) / 256), 256, 0, stream>>>(x, x_h, (int)NF);
    k_tables<<<1, 128, 0, stream>>>(r_w[0], r_b[0], r_w[1], r_b[1], r_w[2], r_b[2],
                                    e0, e1, e2, e3, einit, init0, T);
    hipMemsetAsync(agg, 0, NF * 4, stream);

    // node GEMMs (WMMA)
    k_pre<<<NNODE / 16, 256, 0, stream>>>(x_h, preWt, pre_b, xx_h);
    k_qkvm<<<NNODE / 16, 256, 0, stream>>>(xx_h, msgWt, msg_b[0], msg_b[1], msg_b[2], msg_b[3],
                                           M0g, Qh, Kh, Vh);
    // edge phase (bandwidth-bound, LDS-cached tables)
    k_edge<<<NEDGE / EPB, 256, 0, stream>>>(ei, ea, T, Qh, Kh, Vh, init0, agg);
    // epilogue
    k_hbuild<<<(int)((NF + 255) / 256), 256, 0, stream>>>(M0g, agg, Hh, (int)NF);
    k_post<<<NNODE / 16, 256, 0, stream>>>(Hh, postWt, post_b, x, out);
}
